// SFMGTL_63032940036267
// MI455X (gfx1250) — matrix operand order repaired and checked by reading, not compiled, not run on hardware
//
#include <hip/hip_runtime.h>
#include <hip/hip_bf16.h>
#include <math.h>

// ---------------------------------------------------------------------------
// SFMGTL hierarchical GNN forward for MI455X (gfx1250, wave32, WMMA).
// All dense contractions (x@W, attention aggregation, S^T X, S^T A, (S^T A)S,
// SS^T for link loss, temporal pooling) run on v_wmma_f32_16x16x32_bf16.
// Attention softmax is fused into the WMMA A-fragment (flash-style, 2-pass).
// ---------------------------------------------------------------------------

typedef __bf16 bf16_t;
typedef __attribute__((ext_vector_type(16))) __bf16 v16bf;
typedef __attribute__((ext_vector_type(8)))  float  v8f;

union AFrag { v16bf v; float4 f4[2]; __bf16 e[16]; };

// Problem constants
constexpr int BB   = 4;
constexpr int NN   = 2048;
constexpr int FF   = 64;
constexpr int TT   = 12;
constexpr int HID  = 128;
constexpr int C1   = 100;
constexpr int C2   = 10;
constexpr int C1P  = 112;   // pad C1 to 16
constexpr int M2P  = 112;   // level-2 node rows padded
constexpr int K2P  = 128;   // level-2 node K dim padded to 32
constexpr int TP   = 16;    // pad T

// d_out flat layout: loss, zone_x, zone_temp^T, semantic_x, semantic_temp, qs
constexpr int OUT_ZX = 1;
constexpr int OUT_ZT = OUT_ZX + BB*C1*FF;        // + 25600
constexpr int OUT_SX = OUT_ZT + BB*C1*TT;        // + 4800
constexpr int OUT_ST = OUT_SX + BB*FF;           // + 256
constexpr int OUT_QS = OUT_ST + BB*C2;           // + 40

// ---------------------------------------------------------------------------
// Utility kernels
// ---------------------------------------------------------------------------
__global__ void k_zero(float* p, int n) {
  int i = blockIdx.x*256 + threadIdx.x;
  if (i < n) p[i] = 0.f;
}

// out[r,c] = (r<vr && c<vc) ? in[r*ldin+c] : 0   (fp32 -> bf16, zero padded)
__global__ void k_pack_rm(const float* __restrict__ in, long sIn, int ldin,
                          bf16_t* __restrict__ out, long sOut,
                          int R, int Cc, int vr, int vc) {
  long idx = (long)blockIdx.x*256 + threadIdx.x;
  long tot = (long)R*Cc; if (idx >= tot) return;
  int r = (int)(idx / Cc), c = (int)(idx % Cc);
  float v = (r < vr && c < vc) ? in[(long)blockIdx.y*sIn + (long)r*ldin + c] : 0.f;
  out[(long)blockIdx.y*sOut + idx] = (bf16_t)v;
}

// out[r,c] = (r<vr && c<vc) ? in[c*ldin+r] : 0   (transpose pack)
__global__ void k_pack_tr(const float* __restrict__ in, long sIn, int ldin,
                          bf16_t* __restrict__ out, long sOut,
                          int R, int Cc, int vr, int vc) {
  long idx = (long)blockIdx.x*256 + threadIdx.x;
  long tot = (long)R*Cc; if (idx >= tot) return;
  int r = (int)(idx / Cc), c = (int)(idx % Cc);
  float v = (r < vr && c < vc) ? in[(long)blockIdx.y*sIn + (long)c*ldin + r] : 0.f;
  out[(long)blockIdx.y*sOut + idx] = (bf16_t)v;
}

// ---------------------------------------------------------------------------
// Generic batched GEMM: C[Mp,Np] = A[Mp,Kp] * Bt[Np,Kp]^T  (bf16 in, f32 out)
// One wave per 16x16 output tile; A/B fragments per ISA 7.12.2 layout.
// ---------------------------------------------------------------------------
__global__ void __launch_bounds__(32)
k_gemm(const bf16_t* __restrict__ A, long sA, int lda,
       const bf16_t* __restrict__ Bt, long sB, int ldb,
       float* __restrict__ C, long sC, int ldc, int ksteps) {
  const int tm = blockIdx.x, tn = blockIdx.y, b = blockIdx.z;
  const int lane = threadIdx.x; const bool hiH = lane >= 16; const int l = lane & 15;
  const bf16_t* arow = A  + (long)b*sA + (long)(tm*16 + l)*lda;
  const bf16_t* brow = Bt + (long)b*sB + (long)(tn*16 + l)*ldb;
  v8f acc; for (int e = 0; e < 8; ++e) acc[e] = 0.f;
  for (int ks = 0; ks < ksteps; ++ks) {
    const int k0 = ks*32;
    const int ab = k0 + (hiH ? 8 : 0);
    const int bb = k0 + (hiH ? 8 : 0);
    AFrag fa, fb;
    fa.f4[0] = *(const float4*)(arow + ab);
    fa.f4[1] = *(const float4*)(arow + ab + 16);
    fb.f4[0] = *(const float4*)(brow + bb);
    fb.f4[1] = *(const float4*)(brow + bb + 16);
    __builtin_prefetch(arow + k0 + 64, 0, 1);   // global_prefetch_b8
    acc = __builtin_amdgcn_wmma_f32_16x16x32_bf16(false, fa.v, false, fb.v,
                                                  (short)0, acc, false, false);
  }
  float* crow = C + (long)b*sC;
  const int col = tn*16 + l;
  #pragma unroll
  for (int v = 0; v < 8; ++v)
    crow[(long)(tm*16 + v + (hiH ? 8 : 0))*ldc + col] = acc[v];
}

// ---------------------------------------------------------------------------
// GAT: per-node attention coefficients a_src/a_dst (dot over features)
// ---------------------------------------------------------------------------
__global__ void k_gat_prep(const float* __restrict__ hbuf, long sH, int ldh,
                           const float* __restrict__ att_src,
                           const float* __restrict__ att_dst,
                           float* __restrict__ asrc, float* __restrict__ adst,
                           int Mp, int C, int statN) {
  int idx = blockIdx.x*blockDim.x + threadIdx.x;
  if (idx >= statN*2) return;
  int n = idx >> 1, h = idx & 1, b = blockIdx.y;
  float ss = 0.f, sd = 0.f;
  if (n < Mp) {
    const float* row = hbuf + (long)b*sH + (long)n*ldh + h*C;
    for (int c = 0; c < C; ++c) { float v = row[c]; ss += v*att_src[h*C + c]; sd += v*att_dst[h*C + c]; }
  }
  long o = ((long)b*2 + h)*statN + n;
  asrc[o] = ss; adst[o] = sd;
}

// ---------------------------------------------------------------------------
// GAT softmax row statistics: max / 1/sum of exp(leaky(a_src[j]+a_dst[i]))
// over masked neighbors (adj!=0 or self loop j==i).
// ---------------------------------------------------------------------------
__global__ void k_rowstats(const float* __restrict__ asrc, const float* __restrict__ adst,
                           const float* __restrict__ adj, long sAdj, int ldadj,
                           int nvalid, int statN,
                           float* __restrict__ rmax, float* __restrict__ rinv) {
  const int i = blockIdx.x; const int bh = blockIdx.y; const int b = bh >> 1;
  const float* as  = asrc + (long)bh*statN;
  const float adi  = adst[(long)bh*statN + i];
  const float* arow = adj + (long)b*sAdj + (long)i*ldadj;
  __shared__ float red[128];
  float m = -3.0e38f;
  for (int j = threadIdx.x; j < nvalid; j += 128) {
    if (j == i || arow[j] != 0.f) {
      float e = as[j] + adi; e = e > 0.f ? e : e*0.2f;
      m = fmaxf(m, e);
    }
  }
  red[threadIdx.x] = m; __syncthreads();
  for (int o = 64; o > 0; o >>= 1) { if (threadIdx.x < o) red[threadIdx.x] = fmaxf(red[threadIdx.x], red[threadIdx.x+o]); __syncthreads(); }
  m = red[0]; __syncthreads();
  float s = 0.f;
  for (int j = threadIdx.x; j < nvalid; j += 128) {
    if (j == i || arow[j] != 0.f) {
      float e = as[j] + adi; e = e > 0.f ? e : e*0.2f;
      s += __expf(e - m);
    }
  }
  red[threadIdx.x] = s; __syncthreads();
  for (int o = 64; o > 0; o >>= 1) { if (threadIdx.x < o) red[threadIdx.x] += red[threadIdx.x+o]; __syncthreads(); }
  if (threadIdx.x == 0) {
    float S = red[0];
    long o = (long)bh*statN + i;
    rmax[o] = (S > 0.f) ? m : 0.f;
    rinv[o] = (S > 0.f) ? 1.f/S : 0.f;
  }
}

// ---------------------------------------------------------------------------
// Fused attention aggregation: out[i,c] = 0.5*sum_h sum_j softmax_j(alpha)*h[j,h,c] + bias[c]
// One wave per 16 output rows; softmax weights built in registers, fed as the
// WMMA A fragment; hT (bf16, [H,Cp,Kp]) supplies B fragments contiguously.
// ---------------------------------------------------------------------------
template<int NT>
__global__ void __launch_bounds__(32)
k_attn(const float* __restrict__ asrc, const float* __restrict__ adst,
       const float* __restrict__ rmax, const float* __restrict__ rinv,
       const float* __restrict__ adj, long sAdj, int ldadj,
       int nvalid, int Kp, int statN,
       const bf16_t* __restrict__ hT, long sHTb, long sHTh, int ldhT,
       const float* __restrict__ bias, int cvalid,
       float* __restrict__ out, long sOut, int ldo) {
  const int tm = blockIdx.x, b = blockIdx.y;
  const int lane = threadIdx.x; const bool hiH = lane >= 16; const int l = lane & 15;
  const int i = tm*16 + l;
  v8f acc[NT];
  for (int t = 0; t < NT; ++t) for (int e = 0; e < 8; ++e) acc[t][e] = 0.f;
  const float* arow = adj + (long)b*sAdj + (long)i*ldadj;
  for (int h = 0; h < 2; ++h) {
    const long bh = (long)b*2 + h;
    const float* as = asrc + bh*statN;
    const float adi = adst[bh*statN + i];
    const float rm  = rmax[bh*statN + i];
    const float ri  = rinv[bh*statN + i];
    const bf16_t* hTh = hT + (long)b*sHTb + (long)h*sHTh;
    for (int k0 = 0; k0 < Kp; k0 += 32) {
      const int ab = k0 + (hiH ? 8 : 0);
      AFrag fa;
      #pragma unroll
      for (int u = 0; u < 16; ++u) {
        const int j = ab + (u < 8 ? u : u + 8);
        float e = as[j] + adi;
        e = e > 0.f ? e : e*0.2f;                  // leaky_relu(0.2)
        const float av = (j < nvalid) ? arow[j] : 0.f;
        const bool valid = (j < nvalid) && (j == i || av != 0.f);
        const float w = __expf(e - rm)*ri;         // softmax weight
        fa.e[u] = (bf16_t)(valid ? w : 0.f);
      }
      const int bb = k0 + (hiH ? 8 : 0);
      #pragma unroll
      for (int t = 0; t < NT; ++t) {
        const bf16_t* brow = hTh + (long)(t*16 + l)*ldhT + bb;
        AFrag fb;
        fb.f4[0] = *(const float4*)(brow);
        fb.f4[1] = *(const float4*)(brow + 16);
        acc[t] = __builtin_amdgcn_wmma_f32_16x16x32_bf16(false, fa.v, false, fb.v,
                                                         (short)0, acc[t], false, false);
      }
    }
  }
  #pragma unroll
  for (int t = 0; t < NT; ++t) {
    const int c = t*16 + l;
    const float bi = (c < cvalid) ? bias[c] : 0.f;
    #pragma unroll
    for (int v = 0; v < 8; ++v) {
      const int r = tm*16 + v + (hiH ? 8 : 0);
      out[(long)b*sOut + (long)r*ldo + c] = 0.5f*acc[t][v] + bi;  // head mean + bias
    }
  }
}

// ---------------------------------------------------------------------------
// GraphNorm over the node dimension (per batch, per feature, in place)
// ---------------------------------------------------------------------------
__global__ void k_gnorm(float* __restrict__ x, long sX, int ldx, int nvalid,
                        const float* __restrict__ w, const float* __restrict__ bparm,
                        const float* __restrict__ ms) {
  const int f = blockIdx.x, b = blockIdx.y;
  float* base = x + (long)b*sX + f;
  __shared__ float red[256];
  float s = 0.f;
  for (int n = threadIdx.x; n < nvalid; n += 256) s += base[(long)n*ldx];
  red[threadIdx.x] = s; __syncthreads();
  for (int o = 128; o > 0; o >>= 1) { if (threadIdx.x < o) red[threadIdx.x] += red[threadIdx.x+o]; __syncthreads(); }
  const float mean = red[0]/nvalid; __syncthreads();
  const float msf = ms[f];
  float ss = 0.f;
  for (int n = threadIdx.x; n < nvalid; n += 256) {
    float ov = base[(long)n*ldx] - mean*msf;
    base[(long)n*ldx] = ov;
    ss += ov*ov;
  }
  red[threadIdx.x] = ss; __syncthreads();
  for (int o = 128; o > 0; o >>= 1) { if (threadIdx.x < o) red[threadIdx.x] += red[threadIdx.x+o]; __syncthreads(); }
  const float var = red[0]/nvalid;
  const float sc  = w[f]*rsqrtf(var + 1e-5f);
  const float bc  = bparm[f];
  __syncthreads();
  for (int n = threadIdx.x; n < nvalid; n += 256) base[(long)n*ldx] = base[(long)n*ldx]*sc + bc;
}

// ---------------------------------------------------------------------------
// Row softmax + entropy-loss accumulation
// ---------------------------------------------------------------------------
__global__ void k_softmax_ent(const float* __restrict__ in, long sIn, int ldin,
                              float* __restrict__ sout, long sO, int ldo,
                              int cvalid, float* __restrict__ ent_accum, float ent_scale) {
  const int r = blockIdx.x, b = blockIdx.y;
  const float* row = in + (long)b*sIn + (long)r*ldin;
  float* orow = sout + (long)b*sO + (long)r*ldo;
  __shared__ float red[128];
  float m = -3e38f;
  for (int c = threadIdx.x; c < cvalid; c += 128) m = fmaxf(m, row[c]);
  red[threadIdx.x] = m; __syncthreads();
  for (int o = 64; o > 0; o >>= 1) { if (threadIdx.x < o) red[threadIdx.x] = fmaxf(red[threadIdx.x], red[threadIdx.x+o]); __syncthreads(); }
  m = red[0]; __syncthreads();
  float s = 0.f;
  for (int c = threadIdx.x; c < cvalid; c += 128) s += __expf(row[c] - m);
  red[threadIdx.x] = s; __syncthreads();
  for (int o = 64; o > 0; o >>= 1) { if (threadIdx.x < o) red[threadIdx.x] += red[threadIdx.x+o]; __syncthreads(); }
  const float inv = 1.f/red[0]; __syncthreads();
  float ent = 0.f;
  for (int c = threadIdx.x; c < cvalid; c += 128) {
    float p = __expf(row[c] - m)*inv;
    orow[c] = p;
    ent -= p*__logf(p + 1e-15f);
  }
  red[threadIdx.x] = ent; __syncthreads();
  for (int o = 64; o > 0; o >>= 1) { if (threadIdx.x < o) red[threadIdx.x] += red[threadIdx.x+o]; __syncthreads(); }
  if (threadIdx.x == 0) atomicAdd(ent_accum, red[0]*ent_scale);
}

// ---------------------------------------------------------------------------
// Fused link loss: sum (adj - S S^T)^2 without materializing S S^T.
// One wave computes a 16x16 SS^T tile with WMMA, reduces in-register.
// ---------------------------------------------------------------------------
__global__ void __launch_bounds__(32)
k_link(const bf16_t* __restrict__ sb, long sS, int lds_,
       const float* __restrict__ adj, long sAdj, int ldadj,
       int nvalid, int ksteps, float* __restrict__ accum) {
  const int tm = blockIdx.x, tn = blockIdx.y, b = blockIdx.z;
  const int lane = threadIdx.x; const bool hiH = lane >= 16; const int l = lane & 15;
  const bf16_t* arow = sb + (long)b*sS + (long)(tm*16 + l)*lds_;
  const bf16_t* brow = sb + (long)b*sS + (long)(tn*16 + l)*lds_;
  v8f acc; for (int e = 0; e < 8; ++e) acc[e] = 0.f;
  for (int ks = 0; ks < ksteps; ++ks) {
    const int k0 = ks*32;
    AFrag fa, fb;
    fa.f4[0] = *(const float4*)(arow + k0 + (hiH ? 8 : 0));
    fa.f4[1] = *(const float4*)(arow + k0 + (hiH ? 8 : 0) + 16);
    fb.f4[0] = *(const float4*)(brow + k0 + (hiH ? 8 : 0));
    fb.f4[1] = *(const float4*)(brow + k0 + (hiH ? 8 : 0) + 16);
    acc = __builtin_amdgcn_wmma_f32_16x16x32_bf16(false, fa.v, false, fb.v,
                                                  (short)0, acc, false, false);
  }
  float part = 0.f;
  #pragma unroll
  for (int v = 0; v < 8; ++v) {
    const int r = tm*16 + v + (hiH ? 8 : 0), c = tn*16 + l;
    if (r < nvalid && c < nvalid) {
      float d = adj[(long)b*sAdj + (long)r*ldadj + c] - acc[v];
      part += d*d;
    }
  }
  for (int o = 16; o > 0; o >>= 1) part += __shfl_down(part, o, 32);
  if (lane == 0) atomicAdd(accum, part);
}

// ---------------------------------------------------------------------------
// Output assembly kernels
// ---------------------------------------------------------------------------
__global__ void k_out_zone(const float* __restrict__ zx, float* __restrict__ dout) {
  int idx = blockIdx.x*256 + threadIdx.x;
  if (idx >= BB*C1*FF) return;
  int c = idx % FF, r = (idx/FF) % C1, b = idx/(FF*C1);
  float v = zx[((long)b*C1P + r)*FF + c];
  dout[OUT_ZX + idx] = v;
  dout[OUT_QS + ((long)b*(C1+1) + r)*FF + c] = v;
}
__global__ void k_out_semx(const float* __restrict__ semp, float* __restrict__ dout) {
  int idx = blockIdx.x*64 + threadIdx.x;
  if (idx >= BB*FF) return;
  int c = idx % FF, b = idx/FF;
  float s = 0.f;
  for (int r = 0; r < C2; ++r) s += semp[((long)b*TP + r)*FF + c];
  s *= (1.f/C2);
  dout[OUT_SX + idx] = s;
  dout[OUT_QS + ((long)b*(C1+1) + C1)*FF + c] = s;
}
__global__ void k_out_zt(const float* __restrict__ zt, float* __restrict__ dout) {
  int idx = blockIdx.x*256 + threadIdx.x;
  if (idx >= BB*C1*TT) return;
  int t = idx % TT, c = (idx/TT) % C1, b = idx/(TT*C1);
  dout[OUT_ZT + idx] = zt[((long)b*TP + t)*C1P + c];
}
__global__ void k_out_st(const float* __restrict__ s2f, const float* __restrict__ zt,
                         float* __restrict__ dout) {
  int idx = threadIdx.x;
  if (idx >= BB*C2) return;
  int c = idx % C2, b = idx/C2;
  float s = 0.f;
  for (int t = 0; t < TT; ++t) {
    const float* ztr = zt + ((long)b*TP + t)*C1P;
    for (int n = 0; n < C1; ++n) s += s2f[((long)b*C1P + n)*TP + c]*ztr[n];
  }
  dout[OUT_ST + idx] = s*(1.f/TT);
}
__global__ void k_out_loss(const float* __restrict__ acc, float* __restrict__ dout) {
  if (threadIdx.x == 0) {
    float l1 = sqrtf(acc[0]) / ((float)BB*NN*NN);
    float l2 = sqrtf(acc[1]) / ((float)BB*C1*C1);
    dout[0] = l1 + l2 + acc[2] + acc[3];
  }
}

// ---------------------------------------------------------------------------
// Host-side orchestration
// ---------------------------------------------------------------------------
struct GatP { const float *W, *att_dst, *att_src, *bias; };   // jax dict-key order
struct GnP  { const float *bias, *ms, *weight; };
struct PSet { GnP n1, n2; GatP g1, g2; };

struct Scratch {
  float  *acc, *hbuf, *asrc, *adst, *rmax, *rinv, *t1;
  bf16_t *Wt, *hT, *t1b;
};

static void launch_attn(hipStream_t st, int NT, dim3 g,
    const float* asrc, const float* adst, const float* rmax, const float* rinv,
    const float* adj, long sAdj, int ldadj, int nvalid, int Kp, int statN,
    const bf16_t* hT, long sHTb, long sHTh, int ldhT,
    const float* bias, int cvalid, float* out, long sOut, int ldo) {
  switch (NT) {
    case 8: k_attn<8><<<g,32,0,st>>>(asrc,adst,rmax,rinv,adj,sAdj,ldadj,nvalid,Kp,statN,hT,sHTb,sHTh,ldhT,bias,cvalid,out,sOut,ldo); break;
    case 7: k_attn<7><<<g,32,0,st>>>(asrc,adst,rmax,rinv,adj,sAdj,ldadj,nvalid,Kp,statN,hT,sHTb,sHTh,ldhT,bias,cvalid,out,sOut,ldo); break;
    case 4: k_attn<4><<<g,32,0,st>>>(asrc,adst,rmax,rinv,adj,sAdj,ldadj,nvalid,Kp,statN,hT,sHTb,sHTh,ldhT,bias,cvalid,out,sOut,ldo); break;
    default: k_attn<1><<<g,32,0,st>>>(asrc,adst,rmax,rinv,adj,sAdj,ldadj,nvalid,Kp,statN,hT,sHTb,sHTh,ldhT,bias,cvalid,out,sOut,ldo); break;
  }
}

// One GAT layer: h = Ab @ W  ->  a_src/a_dst  ->  row softmax stats  ->
// fused WMMA attention aggregation (+ head mean + bias).
static void gat_layer(hipStream_t st, const GatP& gp,
    const bf16_t* Ab, int lda, long sA,
    int Mp, int fin, int fout, int Nh2p, int Cp, int NT,
    const float* adj, long sAdj, int ldadj, int nvalid, int Kp,
    Scratch& S, float* out, int ldo, long sOut) {
  const int Nh = 2*fout;
  { // W^T pack (shared across batch): Wt[Nh2p, fin]
    long tot = (long)Nh2p*fin;
    k_pack_tr<<<dim3((unsigned)((tot+255)/256),1),256,0,st>>>(gp.W, 0L, Nh, S.Wt, 0L, Nh2p, fin, Nh, fin);
  }
  k_gemm<<<dim3(Mp/16, Nh2p/16, BB),32,0,st>>>(Ab, sA, lda, S.Wt, 0L, fin,
                                               S.hbuf, (long)Mp*Nh2p, Nh2p, fin/32);
  const int statN = Kp;
  k_gat_prep<<<dim3((statN*2 + 255)/256, BB),256,0,st>>>(S.hbuf, (long)Mp*Nh2p, Nh2p,
      gp.att_src, gp.att_dst, S.asrc, S.adst, Mp, fout, statN);
  for (int h = 0; h < 2; ++h) {  // h^T pack: [Cp, Kp] bf16 per head
    long tot = (long)Cp*Kp;
    k_pack_tr<<<dim3((unsigned)((tot+255)/256), BB),256,0,st>>>(S.hbuf + h*fout,
        (long)Mp*Nh2p, Nh2p, S.hT + (long)h*Cp*Kp, (long)2*Cp*Kp, Cp, Kp, fout, nvalid);
  }
  k_rowstats<<<dim3(Mp, BB*2),128,0,st>>>(S.asrc, S.adst, adj, sAdj, ldadj, nvalid, statN, S.rmax, S.rinv);
  launch_attn(st, NT, dim3(Mp/16, BB), S.asrc, S.adst, S.rmax, S.rinv,
              adj, sAdj, ldadj, nvalid, Kp, statN,
              S.hT, (long)2*Cp*Kp, (long)Cp*Kp, Kp, gp.bias, fout, out, sOut, ldo);
}

// GAT -> GraphNorm -> GAT -> GraphNorm
static void run_gnn(hipStream_t st, const PSet& P,
    const bf16_t* xb, int fin, int lda_x,
    const float* adj, long sAdj, int ldadj, int nvalid, int Kp, int Mp,
    int fout2, int Nh2p2, int Cp2, int NT2,
    Scratch& S, float* out, int ldo) {
  gat_layer(st, P.g1, xb, lda_x, (long)Mp*lda_x, Mp, fin, HID, 2*HID, HID, 8,
            adj, sAdj, ldadj, nvalid, Kp, S, S.t1, HID, (long)Mp*HID);
  k_gnorm<<<dim3(HID, BB),256,0,st>>>(S.t1, (long)Mp*HID, HID, nvalid, P.n1.weight, P.n1.bias, P.n1.ms);
  { long tot = (long)Mp*HID;
    k_pack_rm<<<dim3((unsigned)((tot+255)/256), BB),256,0,st>>>(S.t1, (long)Mp*HID, HID,
        S.t1b, (long)Mp*HID, Mp, HID, nvalid, HID);
  }
  gat_layer(st, P.g2, S.t1b, HID, (long)Mp*HID, Mp, HID, fout2, Nh2p2, Cp2, NT2,
            adj, sAdj, ldadj, nvalid, Kp, S, out, ldo, (long)Mp*ldo);
  k_gnorm<<<dim3(fout2, BB),256,0,st>>>(out, (long)Mp*ldo, ldo, nvalid, P.n2.weight, P.n2.bias, P.n2.ms);
}

extern "C" void kernel_launch(void* const* d_in, const int* in_sizes, int n_in,
                              void* d_out, int out_size, void* d_ws, size_t ws_size,
                              hipStream_t stream) {
  (void)in_sizes; (void)n_in; (void)out_size; (void)ws_size;
  auto F = [&](int i) { return (const float*)d_in[i]; };
  const float* x    = F(0);   // [B,N,64]
  const float* adj  = F(1);   // [B,N,N]
  const float* orig = F(2);   // [B,T,N]
  // jax pytree flatten (dict keys sorted): bn1{bias,mean_scale,weight},
  // bn2{...}, gat1{W,att_dst,att_src,bias}, gat2{...}
  auto parse = [&](int b0) {
    PSet p;
    p.n1 = { F(b0+0),  F(b0+1),  F(b0+2)  };
    p.n2 = { F(b0+3),  F(b0+4),  F(b0+5)  };
    p.g1 = { F(b0+6),  F(b0+7),  F(b0+8),  F(b0+9)  };
    p.g2 = { F(b0+10), F(b0+11), F(b0+12), F(b0+13) };
    return p;
  };
  const PSet pp1 = parse(3), pe1 = parse(17), pp2 = parse(31), pe2 = parse(45);
  float* dout = (float*)d_out;

  // ---- workspace carve-out (~75 MB; assumes ws_size is sufficient) ----
  char* wsb = (char*)d_ws; size_t off = 0;
  auto alloc = [&](size_t bytes) { void* p = wsb + off; off = (off + bytes + 255) & ~(size_t)255; return p; };
  float*  acc   = (float*) alloc(4*sizeof(float));                 // L1sq, L2sq, E1, E2
  bf16_t* xb    = (bf16_t*)alloc((size_t)BB*NN*FF*2);
  bf16_t* adjTb = (bf16_t*)alloc((size_t)BB*NN*NN*2);
  bf16_t* Wt    = (bf16_t*)alloc((size_t)32768*2);
  float*  hbuf  = (float*) alloc((size_t)BB*NN*256*4);
  float*  asrc  = (float*) alloc((size_t)BB*2*NN*4);
  float*  adst  = (float*) alloc((size_t)BB*2*NN*4);
  float*  rmax  = (float*) alloc((size_t)BB*2*NN*4);
  float*  rinv  = (float*) alloc((size_t)BB*2*NN*4);
  bf16_t* hT    = (bf16_t*)alloc((size_t)BB*2*HID*NN*2);
  float*  t1    = (float*) alloc((size_t)BB*NN*HID*4);
  bf16_t* t1b   = (bf16_t*)alloc((size_t)BB*NN*HID*2);
  float*  gs    = (float*) alloc((size_t)BB*NN*C1P*4);             // n2z_s
  float*  gx    = (float*) alloc((size_t)BB*NN*FF*4);              // n2z_x
  float*  s1f   = (float*) alloc((size_t)BB*NN*C1P*4);
  bf16_t* s1b   = (bf16_t*)alloc((size_t)BB*NN*K2P*2);
  bf16_t* s1T   = (bf16_t*)alloc((size_t)BB*C1P*NN*2);
  bf16_t* gxT   = (bf16_t*)alloc((size_t)BB*FF*NN*2);
  float*  zonex = (float*) alloc((size_t)BB*C1P*FF*4);
  float*  tmp1  = (float*) alloc((size_t)BB*C1P*NN*4);
  bf16_t* tmp1b = (bf16_t*)alloc((size_t)BB*C1P*NN*2);
  float*  adjz  = (float*) alloc((size_t)BB*C1P*C1P*4);
  bf16_t* origb = (bf16_t*)alloc((size_t)BB*TP*NN*2);
  float*  zt    = (float*) alloc((size_t)BB*TP*C1P*4);
  bf16_t* zoneb = (bf16_t*)alloc((size_t)BB*C1P*FF*2);
  float*  g2s   = (float*) alloc((size_t)BB*C1P*TP*4);
  float*  g2x   = (float*) alloc((size_t)BB*C1P*FF*4);
  float*  s2f   = (float*) alloc((size_t)BB*C1P*TP*4);
  bf16_t* s2b   = (bf16_t*)alloc((size_t)BB*C1P*32*2);
  bf16_t* s2T   = (bf16_t*)alloc((size_t)BB*TP*K2P*2);
  bf16_t* zxT   = (bf16_t*)alloc((size_t)BB*FF*K2P*2);
  float*  semp  = (float*) alloc((size_t)BB*TP*FF*4);

  Scratch S { acc, hbuf, asrc, adst, rmax, rinv, t1, Wt, hT, t1b };

  k_zero<<<1,32,0,stream>>>(acc, 4);

  // Shared input packs
  k_pack_rm<<<dim3((NN*FF)/256, BB),256,0,stream>>>(x, (long)NN*FF, FF, xb, (long)NN*FF, NN, FF, NN, FF);
  k_pack_tr<<<dim3((unsigned)(((long)NN*NN)/256), BB),256,0,stream>>>(adj, (long)NN*NN, NN, adjTb, (long)NN*NN, NN, NN, NN, NN);
  k_pack_rm<<<dim3((TP*NN)/256, BB),256,0,stream>>>(orig, (long)TT*NN, NN, origb, (long)TP*NN, TP, NN, TT, NN);

  // ---- Level 1: two GNN stacks on the N=2048 graph ----
  run_gnn(stream, pp1, xb, FF, FF, adj, (long)NN*NN, NN, NN, NN, NN,
          C1, 208, C1P, 7, S, gs, C1P);              // n2z_s [B,N,112] (valid 100)
  run_gnn(stream, pe1, xb, FF, FF, adj, (long)NN*NN, NN, NN, NN, NN,
          FF, 128, FF, 4, S, gx, FF);                // n2z_x [B,N,64]

  // ---- DiffPool level 1 ----
  k_softmax_ent<<<dim3(NN, BB),128,0,stream>>>(gs, (long)NN*C1P, C1P, s1f, (long)NN*C1P, C1P,
                                               C1, acc+2, 1.f/((float)BB*NN));
  k_pack_rm<<<dim3(((long)NN*K2P)/256, BB),256,0,stream>>>(s1f, (long)NN*C1P, C1P, s1b, (long)NN*K2P, NN, K2P, NN, C1);
  k_pack_tr<<<dim3(((long)C1P*NN)/256, BB),256,0,stream>>>(s1f, (long)NN*C1P, C1P, s1T, (long)C1P*NN, C1P, NN, C1, NN);
  k_pack_tr<<<dim3(((long)FF*NN)/256, BB),256,0,stream>>>(gx, (long)NN*FF, FF, gxT, (long)FF*NN, FF, NN, FF, NN);
  // zone_x = S^T X
  k_gemm<<<dim3(C1P/16, FF/16, BB),32,0,stream>>>(s1T, (long)C1P*NN, NN, gxT, (long)FF*NN, NN,
                                                  zonex, (long)C1P*FF, FF, NN/32);
  // tmp = S^T A ; adj_z = tmp S
  k_gemm<<<dim3(C1P/16, NN/16, BB),32,0,stream>>>(s1T, (long)C1P*NN, NN, adjTb, (long)NN*NN, NN,
                                                  tmp1, (long)C1P*NN, NN, NN/32);
  k_pack_rm<<<dim3(((long)C1P*NN)/256, BB),256,0,stream>>>(tmp1, (long)C1P*NN, NN, tmp1b, (long)C1P*NN, C1P, NN, C1P, NN);
  k_gemm<<<dim3(C1P/16, C1P/16, BB),32,0,stream>>>(tmp1b, (long)C1P*NN, NN, s1T, (long)C1P*NN, NN,
                                                   adjz, (long)C1P*C1P, C1P, NN/32);
  // link loss 1: sum (adj - S S^T)^2, fused WMMA
  k_link<<<dim3(NN/16, NN/16, BB),32,0,stream>>>(s1b, (long)NN*K2P, K2P, adj, (long)NN*NN, NN,
                                                 NN, K2P/32, acc+0);
  // zone_temp = origin_seq @ S
  k_gemm<<<dim3(TP/16, C1P/16, BB),32,0,stream>>>(origb, (long)TP*NN, NN, s1T, (long)C1P*NN, NN,
                                                  zt, (long)TP*C1P, C1P, NN/32);
  k_pack_rm<<<dim3(((C1P*FF)+255)/256, BB),256,0,stream>>>(zonex, (long)C1P*FF, FF, zoneb, (long)C1P*FF, C1P, FF, C1, FF);

  // ---- Level 2: two GNN stacks on the 100-node zone graph ----
  run_gnn(stream, pp2, zoneb, FF, FF, adjz, (long)C1P*C1P, C1P, C1, K2P, M2P,
          C2, 32, TP, 1, S, g2s, TP);                // z2s_s [B,112,16] (valid 10)
  run_gnn(stream, pe2, zoneb, FF, FF, adjz, (long)C1P*C1P, C1P, C1, K2P, M2P,
          FF, 128, FF, 4, S, g2x, FF);               // z2s_x [B,112,64]

  // ---- DiffPool level 2 / semantic pooling ----
  k_softmax_ent<<<dim3(C1, BB),128,0,stream>>>(g2s, (long)C1P*TP, TP, s2f, (long)C1P*TP, TP,
                                               C2, acc+3, 1.f/((float)BB*C1));
  k_pack_rm<<<dim3(((C1P*32)+255)/256, BB),256,0,stream>>>(s2f, (long)C1P*TP, TP, s2b, (long)C1P*32, C1P, 32, C1, C2);
  k_link<<<dim3(C1P/16, C1P/16, BB),32,0,stream>>>(s2b, (long)C1P*32, 32, adjz, (long)C1P*C1P, C1P,
                                                   C1, 1, acc+1);
  k_pack_tr<<<dim3(((TP*K2P)+255)/256, BB),256,0,stream>>>(s2f, (long)C1P*TP, TP, s2T, (long)TP*K2P, TP, K2P, C2, C1);
  k_pack_tr<<<dim3(((FF*K2P)+255)/256, BB),256,0,stream>>>(g2x, (long)C1P*FF, FF, zxT, (long)FF*K2P, FF, K2P, FF, C1);
  // semantic pre-mean = S2^T Z
  k_gemm<<<dim3(TP/16, FF/16, BB),32,0,stream>>>(s2T, (long)TP*K2P, K2P, zxT, (long)FF*K2P, K2P,
                                                 semp, (long)TP*FF, FF, K2P/32);

  // ---- Output assembly ----
  k_out_zone<<<(BB*C1*FF + 255)/256, 256, 0, stream>>>(zonex, dout);
  k_out_semx<<<(BB*FF + 63)/64, 64, 0, stream>>>(semp, dout);
  k_out_zt<<<(BB*C1*TT + 255)/256, 256, 0, stream>>>(zt, dout);
  k_out_st<<<1, 64, 0, stream>>>(s2f, zt, dout);
  k_out_loss<<<1, 32, 0, stream>>>(acc, dout);
}
